// TransformerMultiAttn_60885456388282
// MI455X (gfx1250) — matrix operand order repaired and checked by reading, not compiled, not run on hardware
//
#include <hip/hip_runtime.h>
#include <hip/hip_bf16.h>

// ---------------------------------------------------------------------------
// TransformerMultiAttn for MI455X (gfx1250, wave32, WMMA)
//
// out = softmax((x@Wk)(x@Wq)^T / 8) @ (x@Wv) @ Wo + bo    (names per ref bug)
//
// Precision: f16 operands, f32 WMMA accumulation (v_wmma_f32_16x16x32_f16).
// GEMM staging uses GLOBAL_LOAD_ASYNC_TO_LDS (ASYNCcnt) + double buffering.
// ---------------------------------------------------------------------------

typedef _Float16 v16h __attribute__((ext_vector_type(16)));
typedef _Float16 v8h  __attribute__((ext_vector_type(8)));
typedef float    v8f  __attribute__((ext_vector_type(8)));
typedef int      v4i  __attribute__((ext_vector_type(4)));

typedef __attribute__((address_space(1))) v4i* as1_v4i_ptr;
typedef __attribute__((address_space(3))) v4i* as3_v4i_ptr;

union HF16 { v16h v; v8h h[2]; };

#define D_IN   1024
#define D_OUT  1024
#define NHEAD  16
#define HDIM   64
#define SEQ    2048
#define BATCH  2
#define MROWS  (BATCH * SEQ)   // 4096

// Epilogue modes for the GEMM kernel
#define MODE_PERHEAD    0   // f16 out[((b*16+h)*2048+q)*64 + d]
#define MODE_PERHEAD_T  1   // f16 out[((b*16+h)*64+d)*2048 + q]
#define MODE_F32_BIAS   2   // f32 out[gm*1024+gn] = acc + bias[gn]

__device__ __forceinline__ v8f wmma_f16(v16h a, v16h b, v8f c) {
  return __builtin_amdgcn_wmma_f32_16x16x32_f16(
      /*neg_a=*/false, a, /*neg_b=*/false, b,
      /*c_mod=*/(short)0, c, /*reuse_a=*/false, /*reuse_b=*/false);
}

// Async global->LDS copy of 16 bytes (one GLOBAL_LOAD_ASYNC_TO_LDS_B128),
// with a synchronous fallback if the builtin is unavailable.
__device__ __forceinline__ void async_copy16(const _Float16* g, _Float16* l) {
#if __has_builtin(__builtin_amdgcn_global_load_async_to_lds_b128)
  __builtin_amdgcn_global_load_async_to_lds_b128(
      (as1_v4i_ptr)(g),   // generic -> AS1 (global)
      (as3_v4i_ptr)(l),   // generic -> AS3 (LDS) addrspacecast
      /*offset=*/0, /*cpol=*/0);
#else
  *(v8h*)l = *(const v8h*)g;
#endif
}

__device__ __forceinline__ void wait_async0() {
#if __has_builtin(__builtin_amdgcn_global_load_async_to_lds_b128)
#if __has_builtin(__builtin_amdgcn_s_wait_asynccnt)
  __builtin_amdgcn_s_wait_asynccnt(0);
#else
  asm volatile("s_wait_asynccnt 0x0" ::: "memory");
#endif
#endif
}

// A-matrix fragment (16x32, MxK), source row-major with leading dim `ld`
// (halves). ISA layout: lane(0..15)=row, halves j=0..7 -> K=8*hl+j,
// j=8..15 -> K=16+8*hl+(j-8), hl = lane>>4.  Two contiguous 16B runs.
__device__ __forceinline__ v16h load_a_frag(const _Float16* base, int ld, int lane) {
  int r  = lane & 15;
  int hl = lane >> 4;
  const _Float16* p = base + r * ld + 8 * hl;
  HF16 u;
  u.h[0] = *(const v8h*)(p);
  u.h[1] = *(const v8h*)(p + 16);
  return u.v;
}

// B-matrix fragment (32x16, KxN), source stored column-contiguous: column n
// of B is a contiguous row of `base` with leading dim `ld` (halves).
// ISA layout: lane(0..15)=col, b[i] -> K = 8*hl+i (i<8), 16+8*hl+(i-8) (i>=8).  Two 16B runs.
__device__ __forceinline__ v16h load_b_frag(const _Float16* base, int ld, int lane) {
  int c  = lane & 15;
  int hl = lane >> 4;
  const _Float16* p = base + c * ld + 8 * hl;
  HF16 u;
  u.h[0] = *(const v8h*)(p);
  u.h[1] = *(const v8h*)(p + 16);
  return u.v;
}

__device__ __forceinline__ float redmax16(float v) {
  v = fmaxf(v, __shfl_xor(v, 8));
  v = fmaxf(v, __shfl_xor(v, 4));
  v = fmaxf(v, __shfl_xor(v, 2));
  v = fmaxf(v, __shfl_xor(v, 1));
  return v;
}
__device__ __forceinline__ float redsum16(float v) {
  v += __shfl_xor(v, 8);
  v += __shfl_xor(v, 4);
  v += __shfl_xor(v, 2);
  v += __shfl_xor(v, 1);
  return v;
}

// ---------------------------------------------------------------------------
// Conversions
// ---------------------------------------------------------------------------
__global__ void cvt_f32_to_f16(const float* __restrict__ src,
                               _Float16* __restrict__ dst, int n) {
  int i = blockIdx.x * 256 + threadIdx.x;
  if (i < n) dst[i] = (_Float16)src[i];
}

// W (K x N, f32 row-major) -> Wt (N x K, f16)
__global__ void cvt_w_transpose(const float* __restrict__ W,
                                _Float16* __restrict__ Wt) {
  int gid = blockIdx.x * 256 + threadIdx.x;  // 1024*1024 threads
  int k = gid >> 10;
  int n = gid & 1023;
  Wt[n * 1024 + k] = (_Float16)W[k * 1024 + n];
}

// ---------------------------------------------------------------------------
// GEMM: C(4096x1024) = A(4096x1024) * B(1024x1024), B given transposed (NxK).
// Block tile 128x128, BK=32, 8 waves, per-wave 64x32 (4x2 WMMA accums).
// Double-buffered async global->LDS staging.
// ---------------------------------------------------------------------------
template <int MODE>
__global__ __launch_bounds__(256) void gemm_wmma(
    const _Float16* __restrict__ A, const _Float16* __restrict__ Bt,
    float* __restrict__ outf, _Float16* __restrict__ outh,
    const float* __restrict__ bias) {
  __shared__ __align__(32) _Float16 As[2][128][32];
  __shared__ __align__(32) _Float16 Bs[2][128][32];

  const int tid  = threadIdx.x;
  const int wave = tid >> 5;
  const int lane = tid & 31;
  const int m0 = blockIdx.y * 128;
  const int n0 = blockIdx.x * 128;
  const int wm = (wave >> 2) * 64;   // 0 or 64
  const int wn = (wave & 3) * 32;    // 0,32,64,96

  v8f acc[4][2];
#pragma unroll
  for (int mi = 0; mi < 4; ++mi)
#pragma unroll
    for (int ni = 0; ni < 2; ++ni)
      acc[mi][ni] = (v8f){0.f, 0.f, 0.f, 0.f, 0.f, 0.f, 0.f, 0.f};

  const int row = tid >> 1;
  const int seg = tid & 1;

  // stage a 128x32 A tile and 128x32 Bt tile into LDS buffer `buf`
  auto stage = [&](int buf, int k0) {
    const _Float16* ga = A  + (size_t)(m0 + row) * D_IN + k0 + seg * 16;
    const _Float16* gb = Bt + (size_t)(n0 + row) * D_IN + k0 + seg * 16;
    async_copy16(ga,     &As[buf][row][seg * 16]);
    async_copy16(ga + 8, &As[buf][row][seg * 16 + 8]);
    async_copy16(gb,     &Bs[buf][row][seg * 16]);
    async_copy16(gb + 8, &Bs[buf][row][seg * 16 + 8]);
  };

  stage(0, 0);

  for (int k0 = 0, it = 0; k0 < D_IN; k0 += 32, ++it) {
    const int cur = it & 1;
    wait_async0();       // our async writes to As/Bs[cur] landed in LDS
    __syncthreads();     // everyone's writes visible
    if (k0 + 32 < D_IN) stage(cur ^ 1, k0 + 32);  // overlap next tile

    v16h va[4], vb[2];
#pragma unroll
    for (int mi = 0; mi < 4; ++mi)
      va[mi] = load_a_frag(&As[cur][wm + mi * 16][0], 32, lane);
#pragma unroll
    for (int ni = 0; ni < 2; ++ni)
      vb[ni] = load_b_frag(&Bs[cur][wn + ni * 16][0], 32, lane);
#pragma unroll
    for (int mi = 0; mi < 4; ++mi)
#pragma unroll
      for (int ni = 0; ni < 2; ++ni)
        acc[mi][ni] = wmma_f16(va[mi], vb[ni], acc[mi][ni]);
    // No trailing barrier needed: this iteration's ds_loads are dscnt-waited
    // before the WMMAs, and the only LDS writes in flight target the other
    // buffer, which is not read until after the next wait+barrier.
  }

  // Epilogue.  C layout: lane<16 -> rows rofs+0..7 at col (lane&15);
  // lane>=16 -> rows rofs+8.. ; VGPR r = row offset.
  const int c    = lane & 15;
  const int rofs = (lane >> 4) * 8;
#pragma unroll
  for (int mi = 0; mi < 4; ++mi) {
#pragma unroll
    for (int ni = 0; ni < 2; ++ni) {
#pragma unroll
      for (int r = 0; r < 8; ++r) {
        int gm = m0 + wm + mi * 16 + rofs + r;
        int gn = n0 + wn + ni * 16 + c;
        float v = acc[mi][ni][r];
        if (MODE == MODE_PERHEAD) {
          int b = gm >> 11, q = gm & 2047;
          int h = gn >> 6,  d = gn & 63;
          outh[(((size_t)(b * NHEAD + h) * SEQ) + q) * HDIM + d] = (_Float16)v;
        } else if (MODE == MODE_PERHEAD_T) {
          int b = gm >> 11, q = gm & 2047;
          int h = gn >> 6,  d = gn & 63;
          outh[(((size_t)(b * NHEAD + h) * HDIM) + d) * SEQ + q] = (_Float16)v;
        } else {  // MODE_F32_BIAS
          outf[(size_t)gm * D_OUT + gn] = v + bias[gn];
        }
      }
    }
  }
}

// ---------------------------------------------------------------------------
// Flash attention: grid (16 q-tiles, 32 bh).  8 waves/block, 16 q-rows/wave.
// Q: [bh][q][d] f16, K: [bh][k][d] f16, Vt: [bh][d][k] f16.
// ctx out: [(b*2048+q)][h*64+d] f16 (row-major 4096x1024).
// ---------------------------------------------------------------------------
__global__ __launch_bounds__(256) void attn_wmma(
    const _Float16* __restrict__ Q, const _Float16* __restrict__ K,
    const _Float16* __restrict__ Vt, _Float16* __restrict__ ctx) {
  __shared__ __align__(32) _Float16 Plds[8][16][32];

  const int tid  = threadIdx.x;
  const int wave = tid >> 5;
  const int lane = tid & 31;
  const int bh = blockIdx.y;
  const int b  = bh >> 4;
  const int h  = bh & 15;
  const int qbase = blockIdx.x * 128 + wave * 16;

  const _Float16* Qbh  = Q  + (size_t)bh * SEQ * HDIM;
  const _Float16* Kbh  = K  + (size_t)bh * SEQ * HDIM;
  const _Float16* Vtbh = Vt + (size_t)bh * HDIM * SEQ;

  // Q fragments for this wave's 16 rows (d-chunks of 32), kept resident.
  v16h qa[2];
#pragma unroll
  for (int dc = 0; dc < 2; ++dc)
    qa[dc] = load_a_frag(Qbh + (size_t)qbase * HDIM + dc * 32, HDIM, lane);

  float m_r[8], l_r[8];
  v8f o[4];
#pragma unroll
  for (int r = 0; r < 8; ++r) { m_r[r] = -1e30f; l_r[r] = 0.f; }
#pragma unroll
  for (int dc = 0; dc < 4; ++dc)
    o[dc] = (v8f){0.f, 0.f, 0.f, 0.f, 0.f, 0.f, 0.f, 0.f};

  const int c    = lane & 15;
  const int rofs = (lane >> 4) * 8;
  const float scale = 0.125f;  // 1/sqrt(64)

  for (int k0 = 0; k0 < SEQ; k0 += 32) {
    // S tile: 16 q-rows x 32 k-cols, accumulated over d=64
    v8f s0 = (v8f){0.f, 0.f, 0.f, 0.f, 0.f, 0.f, 0.f, 0.f};
    v8f s1 = s0;
#pragma unroll
    for (int dc = 0; dc < 2; ++dc) {
      v16h b0 = load_b_frag(Kbh + (size_t)k0 * HDIM + dc * 32, HDIM, lane);
      v16h b1 = load_b_frag(Kbh + (size_t)(k0 + 16) * HDIM + dc * 32, HDIM, lane);
      s0 = wmma_f16(qa[dc], b0, s0);
      s1 = wmma_f16(qa[dc], b1, s1);
    }

    // online softmax (rows striped: VGPR r = row rofs+r, col = c / 16+c)
    float mnew[8], alpha[8];
#pragma unroll
    for (int r = 0; r < 8; ++r) {
      s0[r] *= scale;
      s1[r] *= scale;
      float mc = redmax16(fmaxf(s0[r], s1[r]));
      mnew[r]  = fmaxf(m_r[r], mc);
      alpha[r] = __expf(m_r[r] - mnew[r]);
      float p0 = __expf(s0[r] - mnew[r]);
      float p1 = __expf(s1[r] - mnew[r]);
      s0[r] = p0;
      s1[r] = p1;
      float rs = redsum16(p0 + p1);
      l_r[r] = l_r[r] * alpha[r] + rs;
      m_r[r] = mnew[r];
    }
#pragma unroll
    for (int dc = 0; dc < 4; ++dc)
#pragma unroll
      for (int r = 0; r < 8; ++r)
        o[dc][r] *= alpha[r];

    // stage P (f16) through LDS to re-layout accum -> A-fragment
#pragma unroll
    for (int r = 0; r < 8; ++r) {
      Plds[wave][rofs + r][c]      = (_Float16)s0[r];
      Plds[wave][rofs + r][16 + c] = (_Float16)s1[r];
    }
    __syncthreads();
    v16h pa = load_a_frag(&Plds[wave][0][0], 32, lane);

    // O += P(16x32) * V(32x64)
#pragma unroll
    for (int dc = 0; dc < 4; ++dc) {
      v16h vb = load_b_frag(Vtbh + (size_t)(dc * 16) * SEQ + k0, SEQ, lane);
      o[dc] = wmma_f16(pa, vb, o[dc]);
    }
    __syncthreads();
  }

  // normalize and store ctx
#pragma unroll
  for (int r = 0; r < 8; ++r) {
    float inv = 1.0f / l_r[r];
    int q = qbase + rofs + r;
#pragma unroll
    for (int dc = 0; dc < 4; ++dc) {
      int d = dc * 16 + c;
      ctx[((size_t)(b * SEQ + q)) * D_OUT + h * HDIM + d] =
          (_Float16)(o[dc][r] * inv);
    }
  }
}

// ---------------------------------------------------------------------------
// launch
// ---------------------------------------------------------------------------
extern "C" void kernel_launch(void* const* d_in, const int* in_sizes, int n_in,
                              void* d_out, int out_size, void* d_ws, size_t ws_size,
                              hipStream_t stream) {
  const float* x  = (const float*)d_in[0];
  const float* Wq = (const float*)d_in[1];
  const float* Wk = (const float*)d_in[2];
  const float* Wv = (const float*)d_in[3];
  const float* Wo = (const float*)d_in[4];
  const float* bo = (const float*)d_in[5];
  float* out = (float*)d_out;

  const size_t SZ_X = (size_t)MROWS * D_IN;   // 4.19M halves
  const size_t SZ_W = (size_t)D_IN * D_OUT;   // 1.05M halves

  _Float16* x16  = (_Float16*)d_ws;
  _Float16* wqT  = x16 + SZ_X;
  _Float16* wkT  = wqT + SZ_W;
  _Float16* wvT  = wkT + SZ_W;
  _Float16* woT  = wvT + SZ_W;
  _Float16* Qb   = woT + SZ_W;   // queries = x@Wk, per-head
  _Float16* Kb   = Qb + SZ_X;    // keys    = x@Wq, per-head
  _Float16* Vtb  = Kb + SZ_X;    // values  = x@Wv, per-head transposed
  _Float16* ctx  = Vtb + SZ_X;   // attention output, row-major 4096x1024

  // 1) f16 conversions (+ weight transposes)
  cvt_f32_to_f16<<<(int)(SZ_X / 256), 256, 0, stream>>>(x, x16, (int)SZ_X);
  cvt_w_transpose<<<(int)(SZ_W / 256), 256, 0, stream>>>(Wq, wqT);
  cvt_w_transpose<<<(int)(SZ_W / 256), 256, 0, stream>>>(Wk, wkT);
  cvt_w_transpose<<<(int)(SZ_W / 256), 256, 0, stream>>>(Wv, wvT);
  cvt_w_transpose<<<(int)(SZ_W / 256), 256, 0, stream>>>(Wo, woT);

  dim3 gg(D_OUT / 128, MROWS / 128);  // (8, 32)
  // 2) projections (note swapped names per reference)
  gemm_wmma<MODE_PERHEAD><<<gg, 256, 0, stream>>>(x16, wkT, nullptr, Qb, nullptr);
  gemm_wmma<MODE_PERHEAD><<<gg, 256, 0, stream>>>(x16, wqT, nullptr, Kb, nullptr);
  gemm_wmma<MODE_PERHEAD_T><<<gg, 256, 0, stream>>>(x16, wvT, nullptr, Vtb, nullptr);

  // 3) attention: grid (q-tiles, b*h)
  attn_wmma<<<dim3(SEQ / 128, BATCH * NHEAD), 256, 0, stream>>>(Qb, Kb, Vtb, ctx);

  // 4) output projection + bias -> f32 d_out
  gemm_wmma<MODE_F32_BIAS><<<gg, 256, 0, stream>>>(ctx, woT, out, nullptr, bo);
}
